// instanceMask_12249246728240
// MI455X (gfx1250) — compile-verified, matrix operand order checked by reading, not run on hardware
//
#include <hip/hip_runtime.h>
#include <hip/hip_bf16.h>

// ---------------------------------------------------------------------------
// Problem constants (from reference): B=4, N=32, C=256, H=W=64, P=729, INS=128
// ---------------------------------------------------------------------------
#define BATCH   4
#define NINST   32
#define CCH     256
#define HDIM    64
#define WDIM    64
#define HW      4096
#define PPTS    729
#define PPAD    736          // 46 * 16
#define PT_CNT  46
#define INS     128

typedef __attribute__((ext_vector_type(16))) __bf16 bf16x16;
typedef __attribute__((ext_vector_type(8)))  float  f32x8;

// Load one lane's 16 bf16 elements of a WMMA A/B fragment.
// Caller points p at row_base + (lane>>4)*8; pattern is k = kb+{0..7} and
// kb+16+{0..7} per the CDNA5 16-bit A/B VGPR layout.
__device__ __forceinline__ bf16x16 load_frag16(const __bf16* p) {
  bf16x16 v;
#pragma unroll
  for (int j = 0; j < 8; ++j) v[j] = p[j];
#pragma unroll
  for (int j = 0; j < 8; ++j) v[j + 8] = p[j + 16];
  return v;
}

// ---------------------------------------------------------------------------
// 1) f32 -> bf16 copy of the feature map f (B,C,H,W)
// ---------------------------------------------------------------------------
__global__ void cvt_f_bf16(const float* __restrict__ src, __bf16* __restrict__ dst, int n) {
  int i = blockIdx.x * 256 + threadIdx.x;
  if (i < n) dst[i] = (__bf16)src[i];
}

// ---------------------------------------------------------------------------
// 2) weight repack: w(Cout,CinTot,3,3) f32 -> wbf[kpos][co][ci] bf16
//    (ciOff lets layer 0 skip the temp_ns channel, handled in epilogue)
// ---------------------------------------------------------------------------
__global__ void cvt_w_bf16(const float* __restrict__ w, __bf16* __restrict__ out,
                           int Cout, int CinSel, int CinTot, int ciOff) {
  int i = blockIdx.x * 256 + threadIdx.x;
  int total = 9 * Cout * CinSel;
  if (i >= total) return;
  int ci = i % CinSel;
  int co = (i / CinSel) % Cout;
  int kp = i / (CinSel * Cout);
  out[i] = (__bf16)w[(co * CinTot + (ci + ciOff)) * 9 + kp];
}

// ---------------------------------------------------------------------------
// 3) grid_sample (bilinear, zeros padding, align_corners=False)
//    -> sampled[n][p][c] bf16, P padded to 736 with zero rows
// ---------------------------------------------------------------------------
__global__ void grid_sample_k(const float* __restrict__ f,
                              const float* __restrict__ locations,
                              const int* __restrict__ bidx,
                              __bf16* __restrict__ sampled) {
  long long idx = (long long)blockIdx.x * 256 + threadIdx.x;
  const long long total = (long long)NINST * PPAD * CCH;
  if (idx >= total) return;
  int c = (int)(idx & (CCH - 1));
  int p = (int)((idx >> 8) % PPAD);
  int n = (int)(idx / ((long long)PPAD * CCH));
  __bf16 ov = (__bf16)0.f;
  if (p < PPTS) {
    float ly = locations[(n * PPTS + p) * 2 + 0];
    float lx = locations[(n * PPTS + p) * 2 + 1];
    float gy = 2.f * ly / (HDIM - 1) - 1.f;
    float gx = 2.f * lx / (WDIM - 1) - 1.f;
    float yf = ((gy + 1.f) * HDIM - 1.f) * 0.5f;
    float xf = ((gx + 1.f) * WDIM - 1.f) * 0.5f;
    float x0 = floorf(xf), y0 = floorf(yf);
    float wx1 = xf - x0, wy1 = yf - y0;
    const float* img = f + ((long long)bidx[n] * CCH + c) * HW;
    float acc = 0.f;
#pragma unroll
    for (int dy = 0; dy < 2; ++dy)
#pragma unroll
      for (int dx = 0; dx < 2; ++dx) {
        int xi = (int)x0 + dx, yi = (int)y0 + dy;
        float wt = (dx ? wx1 : 1.f - wx1) * (dy ? wy1 : 1.f - wy1);
        if (xi >= 0 && xi < WDIM && yi >= 0 && yi < HDIM) acc += wt * img[yi * WDIM + xi];
      }
    ov = (__bf16)acc;
  }
  sampled[idx] = ov;
}

// ---------------------------------------------------------------------------
// 4) fused correlation: temp_ns[n][q] = sum_p sigmoid(<sampled[n,:,p],f[n,:,q]>)
//                                       * exp(-d2(p,q)/25)
//    WMMA bf16 GEMM (M=p tile, N=q tile, K=256) + fused branchless epilogue.
//    block = 128 thr (4 waves), covers 64 q columns of one instance.
// ---------------------------------------------------------------------------
__global__ __launch_bounds__(128)
void corr_wmma(const __bf16* __restrict__ f_bf,
               const __bf16* __restrict__ sampled,
               const float* __restrict__ locations,
               const int* __restrict__ bidx,
               float* __restrict__ temp_ns) {
  __shared__ __bf16 Bq[64 * CCH];         // f columns, stored [q][c]  (32 KB)
  __shared__ float locy[PPAD], locx[PPAD];

  const int n  = blockIdx.y;
  const int q0 = blockIdx.x * 64;
  const int tid = threadIdx.x;

  const __bf16* fsrc = f_bf + (long long)bidx[n] * CCH * HW;
  for (int i = tid; i < 64 * CCH; i += 128) {
    int ql = i & 63, c = i >> 6;
    Bq[ql * CCH + c] = fsrc[c * HW + q0 + ql];   // coalesced read, transposed store
  }
  for (int i = tid; i < PPAD; i += 128) {
    float ly = 0.f, lx = 0.f;
    if (i < PPTS) { ly = locations[(n * PPTS + i) * 2 + 0]; lx = locations[(n * PPTS + i) * 2 + 1]; }
    locy[i] = ly; locx[i] = lx;
  }
  __syncthreads();

  const int wave = tid >> 5, lane = tid & 31;
  const int lm = lane & 15, hi = lane >> 4;
  const int qcol = q0 + wave * 16 + lm;
  const float yq = (float)(qcol >> 6), xq = (float)(qcol & 63);

  float colsum = 0.f;
  for (int pt = 0; pt < PT_CNT; ++pt) {
    f32x8 D;
#pragma unroll
    for (int r = 0; r < 8; ++r) D[r] = 0.f;
    const __bf16* arow = sampled + ((long long)n * PPAD + pt * 16 + lm) * CCH + hi * 8;
    const __bf16* brow = &Bq[(wave * 16 + lm) * CCH + hi * 8];
#pragma unroll
    for (int kk = 0; kk < 8; ++kk) {
      bf16x16 A = load_frag16(arow + kk * 32);
      bf16x16 Bv = load_frag16(brow + kk * 32);
      D = __builtin_amdgcn_wmma_f32_16x16x32_bf16(false, A, false, Bv,
                                                  (short)0, D, false, false);
    }
#pragma unroll
    for (int r = 0; r < 8; ++r) {
      int p = pt * 16 + hi * 8 + r;
      float dy = locy[p] - yq, dx = locx[p] - xq;
      float mask = (p < PPTS) ? 1.f : 0.f;                 // v_cndmask, no branch
      float wgt = mask * __expf(-(dy * dy + dx * dx) * 0.04f);
      float sg = __builtin_amdgcn_rcpf(1.f + __expf(-D[r]));  // fast v_rcp_f32
      colsum += sg * wgt;
    }
  }
  colsum += __shfl_xor(colsum, 16, 32);     // combine row-halves (same q column)
  if (lane < 16) temp_ns[(long long)n * HW + qcol] = colsum;
}

// ---------------------------------------------------------------------------
// 5) 3x3 conv as implicit WMMA GEMM. block = 256 thr (8 waves) handles one
//    (n, y) row: 128 cout x 64 x.  Wave w -> cout tile [16w,16w+16), 4 x-tiles.
//    LDS stages 3 input rows x 64-channel chunk x 66 px (x halo zero-padded).
//    FIRST adds bias-like rank-1 temp_ns channel (w0 channel 0) in epilogue.
// ---------------------------------------------------------------------------
template <int CIN, bool FIRST>
__global__ __launch_bounds__(256)
void conv3x3_wmma(const __bf16* __restrict__ in_base,   // f_bf (FIRST) or act
                  const __bf16* __restrict__ wbf,       // [kpos][co][ci(CIN)]
                  const float*  __restrict__ bias,      // [128]
                  const float*  __restrict__ w0_f32,    // (128,257,3,3), FIRST only
                  const float*  __restrict__ temp_ns,   // (n,4096), FIRST only
                  const int*    __restrict__ bidx,      // FIRST only
                  __bf16* __restrict__ out) {           // (n,128,64,64)
  __shared__ __bf16 lds[3 * 68 * 64];   // [row][xi(68)][ci(64)], 26 KB

  const int y = blockIdx.x;
  const int n = blockIdx.y;
  const int tid = threadIdx.x;
  const int wave = tid >> 5, lane = tid & 31;
  const int lm = lane & 15, hi = lane >> 4;
  const int co_base = wave * 16;

  const __bf16* src;
  if constexpr (FIRST) src = in_base + (long long)bidx[n] * CIN * HW;
  else                 src = in_base + (long long)n * CIN * HW;

  // zero the x-halo columns once (xi = 0, 65, 66, 67)
  for (int i = tid; i < 3 * 64 * 4; i += 256) {
    int r = i / 256, rem = i % 256;
    int ci = rem >> 2, bsel = rem & 3;
    int xi = (bsel == 0) ? 0 : 64 + bsel;
    lds[(r * 68 + xi) * 64 + ci] = (__bf16)0.f;
  }

  f32x8 D[4];
#pragma unroll
  for (int xt = 0; xt < 4; ++xt)
#pragma unroll
    for (int r = 0; r < 8; ++r) D[xt][r] = 0.f;

  for (int cc = 0; cc < CIN / 64; ++cc) {
    __syncthreads();
    for (int i = tid; i < 3 * 64 * 64; i += 256) {
      int x = i & 63, ci = (i >> 6) & 63, r = i >> 12;
      int yy = y + r - 1;
      __bf16 bv = (__bf16)0.f;
      if (yy >= 0 && yy < HDIM) bv = src[(cc * 64 + ci) * HW + yy * WDIM + x];
      lds[(r * 68 + (x + 1)) * 64 + ci] = bv;
    }
    __syncthreads();

    // prefetch next chunk's input lines (1 cacheline each) while we compute
    if (cc + 1 < CIN / 64) {
      for (int i = tid; i < 3 * 64; i += 256) {
        int ci = i & 63, r = i >> 6;
        int yy = y + r - 1;
        if (yy >= 0 && yy < HDIM)
          __builtin_prefetch(&src[((cc + 1) * 64 + ci) * HW + yy * WDIM], 0, 1);
      }
    }

#pragma unroll
    for (int kc = 0; kc < 2; ++kc) {
#pragma unroll
      for (int ky = 0; ky < 3; ++ky) {
#pragma unroll
        for (int kx = 0; kx < 3; ++kx) {
          const __bf16* ap = wbf + ((long long)(ky * 3 + kx) * INS + co_base + lm) * CIN
                                 + cc * 64 + kc * 32 + hi * 8;
          bf16x16 A = load_frag16(ap);
#pragma unroll
          for (int xt = 0; xt < 4; ++xt) {
            int xi = xt * 16 + lm + kx;            // x_in + 1, in [0,65]
            const __bf16* bp = &lds[(ky * 68 + xi) * 64 + kc * 32 + hi * 8];
            bf16x16 Bv = load_frag16(bp);
            D[xt] = __builtin_amdgcn_wmma_f32_16x16x32_bf16(false, A, false, Bv,
                                                            (short)0, D[xt], false, false);
          }
        }
      }
    }
  }

  // epilogue: bias (+ temp_ns channel for layer 0), relu, bf16 store
#pragma unroll
  for (int xt = 0; xt < 4; ++xt) {
    int x = xt * 16 + lm;
#pragma unroll
    for (int r = 0; r < 8; ++r) {
      int co = co_base + hi * 8 + r;
      float v = D[xt][r] + bias[co];
      if constexpr (FIRST) {
        float t = 0.f;
#pragma unroll
        for (int ky = 0; ky < 3; ++ky)
#pragma unroll
          for (int kx = 0; kx < 3; ++kx) {
            int yy = y + ky - 1, xx = x + kx - 1;
            if (yy >= 0 && yy < HDIM && xx >= 0 && xx < WDIM)
              t += w0_f32[(co * 257) * 9 + ky * 3 + kx] *
                   temp_ns[(long long)n * HW + yy * WDIM + xx];
          }
        v += t;
      }
      v = fmaxf(v, 0.f);
      out[((long long)n * INS + co) * HW + y * WDIM + x] = (__bf16)v;
    }
  }
}

// ---------------------------------------------------------------------------
// 6) final 1-channel 3x3 conv (0.3 GFLOP) -> d_out f32
// ---------------------------------------------------------------------------
__global__ void score_k(const __bf16* __restrict__ act,
                        const float* __restrict__ sw, const float* __restrict__ sb,
                        float* __restrict__ out) {
  int idx = blockIdx.x * 256 + threadIdx.x;
  if (idx >= NINST * HW) return;
  int q = idx & (HW - 1), n = idx >> 12;
  int y = q >> 6, x = q & 63;
  float acc = sb[0];
  const __bf16* a = act + (long long)n * INS * HW;
  for (int ci = 0; ci < INS; ++ci) {
    const float* wr = sw + ci * 9;
    const __bf16* ap = a + ci * HW;
#pragma unroll
    for (int ky = 0; ky < 3; ++ky) {
      int yy = y + ky - 1;
      if (yy < 0 || yy >= HDIM) continue;
#pragma unroll
      for (int kx = 0; kx < 3; ++kx) {
        int xx = x + kx - 1;
        if (xx < 0 || xx >= WDIM) continue;
        acc += wr[ky * 3 + kx] * (float)ap[yy * WDIM + xx];
      }
    }
  }
  out[idx] = acc;
}

// ---------------------------------------------------------------------------
// host orchestration
// ---------------------------------------------------------------------------
static inline size_t alignup(size_t v) { return (v + 255) & ~(size_t)255; }

extern "C" void kernel_launch(void* const* d_in, const int* in_sizes, int n_in,
                              void* d_out, int out_size, void* d_ws, size_t ws_size,
                              hipStream_t stream) {
  (void)in_sizes; (void)n_in; (void)out_size; (void)ws_size;
  const float* f    = (const float*)d_in[0];
  const float* loc  = (const float*)d_in[1];
  const float* w0   = (const float*)d_in[2];
  const float* b0   = (const float*)d_in[3];
  const float* w1   = (const float*)d_in[4];
  const float* b1   = (const float*)d_in[5];
  const float* w2   = (const float*)d_in[6];
  const float* b2   = (const float*)d_in[7];
  const float* w3   = (const float*)d_in[8];
  const float* b3   = (const float*)d_in[9];
  const float* scw  = (const float*)d_in[10];
  const float* scb  = (const float*)d_in[11];
  const int*   bix  = (const int*)d_in[12];
  float* out = (float*)d_out;

  // workspace carve-up
  char* p = (char*)d_ws;
  __bf16* f_bf  = (__bf16*)p; p += alignup((size_t)BATCH * CCH * HW * 2);      // 8 MB
  __bf16* samp  = (__bf16*)p; p += alignup((size_t)NINST * PPAD * CCH * 2);    // 11.5 MB
  float*  tns   = (float*) p; p += alignup((size_t)NINST * HW * 4);            // 0.5 MB
  __bf16* wb0   = (__bf16*)p; p += alignup((size_t)9 * INS * CCH * 2);
  __bf16* wb1   = (__bf16*)p; p += alignup((size_t)9 * INS * INS * 2);
  __bf16* wb2   = (__bf16*)p; p += alignup((size_t)9 * INS * INS * 2);
  __bf16* wb3   = (__bf16*)p; p += alignup((size_t)9 * INS * INS * 2);
  __bf16* actA  = (__bf16*)p; p += alignup((size_t)NINST * INS * HW * 2);      // 32 MB
  __bf16* actB  = (__bf16*)p; p += alignup((size_t)NINST * INS * HW * 2);      // 32 MB

  // precision conversions / repacks
  {
    int nF = BATCH * CCH * HW;
    cvt_f_bf16<<<(nF + 255) / 256, 256, 0, stream>>>(f, f_bf, nF);
    int n0 = 9 * INS * CCH;
    cvt_w_bf16<<<(n0 + 255) / 256, 256, 0, stream>>>(w0, wb0, INS, CCH, 257, 1);
    int n1 = 9 * INS * INS;
    cvt_w_bf16<<<(n1 + 255) / 256, 256, 0, stream>>>(w1, wb1, INS, INS, INS, 0);
    cvt_w_bf16<<<(n1 + 255) / 256, 256, 0, stream>>>(w2, wb2, INS, INS, INS, 0);
    cvt_w_bf16<<<(n1 + 255) / 256, 256, 0, stream>>>(w3, wb3, INS, INS, INS, 0);
  }

  // bilinear sampling -> sampled (n, 736, 256) bf16
  {
    long long tot = (long long)NINST * PPAD * CCH;
    grid_sample_k<<<(unsigned)((tot + 255) / 256), 256, 0, stream>>>(f, loc, bix, samp);
  }

  // fused correlation -> temp_ns
  corr_wmma<<<dim3(HW / 64, NINST), 128, 0, stream>>>(f_bf, samp, loc, bix, tns);

  // conv stack
  conv3x3_wmma<CCH, true ><<<dim3(HDIM, NINST), 256, 0, stream>>>(f_bf, wb0, b0, w0, tns, bix, actA);
  conv3x3_wmma<INS, false><<<dim3(HDIM, NINST), 256, 0, stream>>>(actA, wb1, b1, nullptr, nullptr, nullptr, actB);
  conv3x3_wmma<INS, false><<<dim3(HDIM, NINST), 256, 0, stream>>>(actB, wb2, b2, nullptr, nullptr, nullptr, actA);
  conv3x3_wmma<INS, false><<<dim3(HDIM, NINST), 256, 0, stream>>>(actA, wb3, b3, nullptr, nullptr, nullptr, actB);

  // final 1-channel score conv
  score_k<<<(NINST * HW + 255) / 256, 256, 0, stream>>>(actB, scw, scb, out);
}
